// BiPointNetConv_35158602285607
// MI455X (gfx1250) — 8-bit matrix operand order repaired and checked by reading, compiled, not run on hardware
//
#include <hip/hip_runtime.h>
#include <stdint.h>

typedef __attribute__((ext_vector_type(8))) int v8i;

#define NPIX   (8 * 32 * 1024)
#define NTILES (NPIX / 16)

__device__ __forceinline__ int fsign(float x) { return (x > 0.f) - (x < 0.f); }
__device__ __forceinline__ int pack4(int a, int b, int c, int d) {
  return (a & 255) | ((b & 255) << 8) | ((c & 255) << 16) | ((d & 255) << 24);
}

template <int COUT, bool FIRST>
__global__ void __launch_bounds__(256)
bi_matmul_kernel(const float* __restrict__ actf,
                 const signed char* __restrict__ acti,
                 const float* __restrict__ W,
                 const float* __restrict__ cA,
                 const float* __restrict__ cB,
                 signed char* __restrict__ yout) {
  __shared__ float sA[64];
  __shared__ float sB[64];
  if (!FIRST) {
    if (threadIdx.x < 64) {
      sA[threadIdx.x] = cA[threadIdx.x];
      sB[threadIdx.x] = cB[threadIdx.x];
    }
    __syncthreads();
  }
  const int lane = threadIdx.x & 31;
  const int wid  = threadIdx.x >> 5;
  const int hi   = lane >> 4;
  const int lx   = lane & 15;

  constexpr int MT = COUT / 16;
  v8i A[MT];
#pragma unroll
  for (int m = 0; m < MT; ++m) {
    const float* wrow = W + (m * 16 + lx) * 64;
#pragma unroll
    for (int v = 0; v < 8; ++v) {
      const int ks = ((v >> 2) << 5) + (hi << 4) + ((v & 3) << 2);
      A[m][v] = pack4(fsign(wrow[ks]), fsign(wrow[ks + 1]),
                      fsign(wrow[ks + 2]), fsign(wrow[ks + 3]));
    }
  }

  const int nwaves = (gridDim.x * blockDim.x) >> 5;
  for (int t = blockIdx.x * (blockDim.x >> 5) + wid; t < NTILES; t += nwaves) {
    const int pix = t * 16 + lx;
    v8i Bm;
    if (FIRST) {
      const int b = pix >> 15;
      const int k = (pix >> 10) & 31;
      const int n = pix & 1023;
      const float* src = actf + (((size_t)((b << 10) | n)) * 32 + (size_t)k) * 64;
#pragma unroll
      for (int g = 0; g < 2; ++g) {
        const int kb = g * 32 + hi * 16;
        const float4* f4 = (const float4*)(src + kb);
#pragma unroll
        for (int w = 0; w < 4; ++w) {
          const float4 q = f4[w];
          Bm[g * 4 + w] = pack4(fsign(q.x), fsign(q.y), fsign(q.z), fsign(q.w));
        }
      }
    } else {
      const uint32_t* row = (const uint32_t*)(acti + (size_t)pix * 64);
#pragma unroll
      for (int g = 0; g < 2; ++g) {
        const int kb = g * 32 + hi * 16;
#pragma unroll
        for (int w = 0; w < 4; ++w) {
          const uint32_t u = row[(kb >> 2) + w];
          int bits[4];
#pragma unroll
          for (int j = 0; j < 4; ++j) {
            const int sv = (int)(signed char)((u >> (8 * j)) & 0xffu);
            const int ki = kb + 4 * w + j;
            bits[j] = (fmaf(sA[ki], (float)sv, sB[ki]) > 0.f) ? 1 : 0;
          }
          Bm[g * 4 + w] = pack4(bits[0], bits[1], bits[2], bits[3]);
        }
      }
    }
    signed char* orow = yout + (size_t)pix * COUT + hi * 8;
#pragma unroll
    for (int m = 0; m < MT; ++m) {
      v8i c = {0, 0, 0, 0, 0, 0, 0, 0};
      c = __builtin_amdgcn_wmma_i32_16x16x64_iu8(true, A[m], true, Bm, c,
                                                 false, false);
      int2 pk;
      pk.x = pack4(c[0], c[1], c[2], c[3]);
      pk.y = pack4(c[4], c[5], c[6], c[7]);
      *(int2*)(orow + m * 16) = pk;
    }
  }
}

template <int C>
__global__ void __launch_bounds__(256)
reduce_kernel(const signed char* __restrict__ y, int* __restrict__ gsum,
              int* __restrict__ gsq) {
  constexpr int G = C / 4;
  constexpr int ROWS = 256 / G;
  const int cg = threadIdx.x & (G - 1);
  const int pr = threadIdx.x / G;
  const uint32_t* yy = (const uint32_t*)y;
  int s[4] = {0, 0, 0, 0};
  int q[4] = {0, 0, 0, 0};
  for (int p = blockIdx.x * ROWS + pr; p < NPIX; p += gridDim.x * ROWS) {
    const uint32_t u = yy[(size_t)p * G + cg];
#pragma unroll
    for (int j = 0; j < 4; ++j) {
      const int v = (int)(signed char)((u >> (8 * j)) & 0xffu);
      s[j] += v;
      q[j] += v * v;
    }
  }
  __shared__ int ls[C];
  __shared__ int lq[C];
  if (threadIdx.x < C) { ls[threadIdx.x] = 0; lq[threadIdx.x] = 0; }
  __syncthreads();
#pragma unroll
  for (int j = 0; j < 4; ++j) {
    atomicAdd(&ls[cg * 4 + j], s[j]);
    atomicAdd(&lq[cg * 4 + j], q[j]);
  }
  __syncthreads();
  if (threadIdx.x < C) {
    atomicAdd(&gsum[threadIdx.x], ls[threadIdx.x]);
    atomicAdd(&gsq[threadIdx.x], lq[threadIdx.x]);
  }
}

__global__ void bncoef_kernel(const int* __restrict__ gsum,
                              const int* __restrict__ gsq,
                              const float* __restrict__ scale,
                              const float* __restrict__ gamma,
                              const float* __restrict__ beta,
                              float* __restrict__ cA, float* __restrict__ cB,
                              int C) {
  const int c = threadIdx.x;
  if (c >= C) return;
  const float invN = 1.0f / (float)NPIX;
  const float m   = (float)gsum[c] * invN;
  const float var = (float)gsq[c] * invN - m * m;
  const float s   = scale[c];
  const float a   = gamma[c] * s * rsqrtf(s * s * var + 1e-5f);
  cA[c] = a;
  cB[c] = beta[c] - a * m;
}

__global__ void zero_kernel(int* __restrict__ p, int n) {
  const int i = blockIdx.x * blockDim.x + threadIdx.x;
  if (i < n) p[i] = 0;
}

__global__ void __launch_bounds__(256)
finalmax_kernel(const signed char* __restrict__ y3,
                const float* __restrict__ cA, const float* __restrict__ cB,
                float* __restrict__ out) {
  const int idx = blockIdx.x * blockDim.x + threadIdx.x;
  const int m  = idx >> 5;
  const int cg = idx & 31;
  const int b = m >> 10;
  const int n = m & 1023;
  const uint32_t* yy = (const uint32_t*)y3;
  const size_t base = ((size_t)b * 32768 + (size_t)n) * 32 + (size_t)cg;
  int mx[4] = {-128, -128, -128, -128};
  int mn[4] = {127, 127, 127, 127};
  for (int k = 0; k < 32; ++k) {
    const uint32_t u = yy[base + (size_t)k * 32768];
#pragma unroll
    for (int j = 0; j < 4; ++j) {
      const int v = (int)(signed char)((u >> (8 * j)) & 0xffu);
      mx[j] = v > mx[j] ? v : mx[j];
      mn[j] = v < mn[j] ? v : mn[j];
    }
  }
#pragma unroll
  for (int j = 0; j < 4; ++j) {
    const int c = cg * 4 + j;
    const float a = cA[c], bb = cB[c];
    const float r = fmaxf(fmaxf(fmaf(a, (float)mx[j], bb),
                                fmaf(a, (float)mn[j], bb)), 0.0f);
    out[(size_t)m * 128 + c] = r;
  }
}

extern "C" void kernel_launch(void* const* d_in, const int* in_sizes, int n_in,
                              void* d_out, int out_size, void* d_ws,
                              size_t ws_size, hipStream_t stream) {
  const float* agg = (const float*)d_in[0];
  const float* W1  = (const float*)d_in[1];
  const float* s1  = (const float*)d_in[2];
  const float* g1  = (const float*)d_in[3];
  const float* b1  = (const float*)d_in[4];
  const float* W2  = (const float*)d_in[5];
  const float* s2  = (const float*)d_in[6];
  const float* g2  = (const float*)d_in[7];
  const float* b2  = (const float*)d_in[8];
  const float* W3  = (const float*)d_in[9];
  const float* s3  = (const float*)d_in[10];
  const float* g3  = (const float*)d_in[11];
  const float* b3  = (const float*)d_in[12];
  float* out = (float*)d_out;

  char* ws = (char*)d_ws;
  size_t off = 0;
  signed char* y1 = (signed char*)(ws + off); off += (size_t)NPIX * 64;
  signed char* y2 = (signed char*)(ws + off); off += (size_t)NPIX * 64;
  signed char* y3 = (signed char*)(ws + off); off += (size_t)NPIX * 128;
  int* sums = (int*)(ws + off); off += 6 * 128 * sizeof(int);
  float* coef = (float*)(ws + off); off += 6 * 128 * sizeof(float);
  if (ws_size < off) return;

  int* sum1 = sums + 0 * 128;  int* sq1 = sums + 1 * 128;
  int* sum2 = sums + 2 * 128;  int* sq2 = sums + 3 * 128;
  int* sum3 = sums + 4 * 128;  int* sq3 = sums + 5 * 128;
  float* cA1 = coef + 0 * 128; float* cB1 = coef + 1 * 128;
  float* cA2 = coef + 2 * 128; float* cB2 = coef + 3 * 128;
  float* cA3 = coef + 4 * 128; float* cB3 = coef + 5 * 128;

  zero_kernel<<<3, 256, 0, stream>>>(sums, 6 * 128);

  bi_matmul_kernel<64, true><<<256, 256, 0, stream>>>(agg, nullptr, W1,
                                                      nullptr, nullptr, y1);
  reduce_kernel<64><<<128, 256, 0, stream>>>(y1, sum1, sq1);
  bncoef_kernel<<<1, 128, 0, stream>>>(sum1, sq1, s1, g1, b1, cA1, cB1, 64);

  bi_matmul_kernel<64, false><<<256, 256, 0, stream>>>(nullptr, y1, W2, cA1,
                                                       cB1, y2);
  reduce_kernel<64><<<128, 256, 0, stream>>>(y2, sum2, sq2);
  bncoef_kernel<<<1, 128, 0, stream>>>(sum2, sq2, s2, g2, b2, cA2, cB2, 64);

  bi_matmul_kernel<128, false><<<256, 256, 0, stream>>>(nullptr, y2, W3, cA2,
                                                        cB2, y3);
  reduce_kernel<128><<<128, 256, 0, stream>>>(y3, sum3, sq3);
  bncoef_kernel<<<1, 128, 0, stream>>>(sum3, sq3, s3, g3, b3, cA3, cB3, 128);

  finalmax_kernel<<<1024, 256, 0, stream>>>(y3, cA3, cB3, out);
}
